// MultiheadAttention_17162689315590
// MI455X (gfx1250) — matrix operand order repaired and checked by reading, not compiled, not run on hardware
//
#include <hip/hip_runtime.h>
#include <hip/hip_bf16.h>

// ---- problem constants (from reference) ----
#define T_DIM 1024
#define B_DIM 8
#define E_DIM 1024
#define H_DIM 16
#define NB_DIM 4
#define TOPK 16
#define HD 64
#define DIN 256
#define ROWS (T_DIM * B_DIM)    // 8192
#define NHEAD (B_DIM * H_DIM)   // 128
#define SCALING 0.125f          // 64^-0.5
#define CHUNK 128               // keys staged per TDM transfer
#define NCHUNK (T_DIM / CHUNK)  // 8

typedef __bf16 v16bf __attribute__((ext_vector_type(16)));
typedef float  v8f   __attribute__((ext_vector_type(8)));
typedef unsigned int uint4v __attribute__((ext_vector_type(4)));
typedef int    int4v  __attribute__((ext_vector_type(4)));
typedef int    int8v  __attribute__((ext_vector_type(8)));

union FragU { uint4v u[2]; v16bf v; };

__device__ __forceinline__ v16bf load_frag(const __bf16* p0, const __bf16* p1) {
    FragU f;
    f.u[0] = *(const uint4v*)p0;
    f.u[1] = *(const uint4v*)p1;
    return f.v;
}

#if __has_builtin(__builtin_amdgcn_tensor_load_to_lds) && \
    __has_builtin(__builtin_amdgcn_s_wait_tensorcnt)
#define HAS_TDM 1
#else
#define HAS_TDM 0
#endif

#if HAS_TDM
// byte offset of a __shared__ object within the wave's LDS aperture
__device__ __forceinline__ unsigned lds_off_of(const void* p) {
    return (unsigned)(unsigned long long)
        (const __attribute__((address_space(3))) void*)p;
}

// Issue a TDM load of `nelem` contiguous 2-byte elements (global -> LDS).
// D# per CDNA5 ISA: 1-D tile, data_size=2B, type=2.
__device__ __forceinline__ void tdm_load_contig(unsigned lds_off,
                                                const void* gsrc,
                                                unsigned nelem) {
    unsigned long long ga = (unsigned long long)(uintptr_t)gsrc;
    uint4v g0;
    g0[0] = 1u;                                   // count=1 (valid descriptor)
    g0[1] = lds_off;                              // lds_addr
    g0[2] = (unsigned)(ga & 0xffffffffu);         // global_addr[31:0]
    g0[3] = (unsigned)((ga >> 32) & 0x01ffffffu)  // global_addr[56:32]
            | 0x80000000u;                        // type=2 (bits 127:126 = 10b)
    int8v g1 = {};
    g1[0] = 0x00010000;                           // data_size=1 (2 bytes)
    g1[1] = (int)((nelem & 0xffffu) << 16);       // tensor_dim0[15:0] @ bit48
    g1[2] = (int)(((nelem >> 16) & 0xffffu)       // tensor_dim0[31:16]
                  | (1u << 16));                  // tensor_dim1 = 1
    g1[3] = (int)((nelem & 0xffffu) << 16);       // tile_dim0 @ bit112
    g1[5] = (int)nelem;                           // tensor_dim0_stride
    int4v gz = {};
#if defined(__clang_major__) && __clang_major__ >= 23
    int8v gz8 = {};
    __builtin_amdgcn_tensor_load_to_lds(g0, g1, gz, gz, gz8, 0);
#else
    __builtin_amdgcn_tensor_load_to_lds(g0, g1, gz, gz, 0);
#endif
}
#endif  // HAS_TDM

// -----------------------------------------------------------------------------
// elementwise fp32 -> bf16 (vectorized by 4)
// -----------------------------------------------------------------------------
__global__ __launch_bounds__(256) void cvt_bf16_kernel(
    const float* __restrict__ in, __bf16* __restrict__ out, int n4) {
    int i = blockIdx.x * 256 + threadIdx.x;
    if (i < n4) {
        float4 v = ((const float4*)in)[i];
        __bf16* o = out + (size_t)i * 4;
        o[0] = (__bf16)v.x; o[1] = (__bf16)v.y;
        o[2] = (__bf16)v.z; o[3] = (__bf16)v.w;
    }
}

// W[n][k][c] fp32 -> Wt[n][c][k] bf16 (so B fragments are contiguous in K)
__global__ __launch_bounds__(256) void wtrans_kernel(
    const float* __restrict__ W, __bf16* __restrict__ Wt) {
    int n = blockIdx.y, k = blockIdx.x, c = threadIdx.x;
    Wt[((size_t)n * DIN + c) * DIN + k] =
        (__bf16)W[((size_t)n * DIN + k) * DIN + c];
}

// -----------------------------------------------------------------------------
// Barrier-free streaming block-diagonal GEMM (bf16 in, f32 accumulate).
// WG = 8 waves, tile 32 rows x 256 cols; wave tile 16x64 (4 WMMA accums).
// OUT_MODE 0: bf16 head-major [head][t][64].  OUT_MODE 1: fp32 (T,B,E).
// -----------------------------------------------------------------------------
template <int OUT_MODE>
__global__ __launch_bounds__(256) void proj_kernel(
    const __bf16* __restrict__ xb,  // (ROWS, E) bf16
    const __bf16* __restrict__ Wt,  // (NB, DIN, DIN) bf16, [n][col][k]
    const float* __restrict__ bias, float scale,
    __bf16* __restrict__ out_bf, float* __restrict__ out_f) {
    const int n    = blockIdx.y;
    const int row0 = blockIdx.x * 32;
    const int tid  = threadIdx.x;
    const int lane = tid & 31;
    const int wave = tid >> 5;
    const int wm   = wave >> 2;   // 0..1
    const int wn   = wave & 3;    // 0..3
    const int M    = lane & 15;
    const int hi   = lane >> 4;

    v8f acc[4];
    #pragma unroll
    for (int i = 0; i < 4; i++) acc[i] = {};

    const __bf16* arow = xb + (size_t)(row0 + wm * 16 + M) * E_DIM + n * DIN;
    const __bf16* wbase = Wt + (size_t)n * DIN * DIN;

    #pragma unroll
    for (int k0 = 0; k0 < DIN; k0 += 32) {
        // A frag: lane row M, run [hi*16..+16)
        v16bf a = load_frag(arow + k0 + hi * 16, arow + k0 + hi * 16 + 8);
        #pragma unroll
        for (int nn = 0; nn < 4; nn++) {
            const __bf16* wc =
                wbase + (size_t)(wn * 64 + nn * 16 + M) * DIN + k0 + hi * 16;
            v16bf b = load_frag(wc, wc + 8);
            acc[nn] = __builtin_amdgcn_wmma_f32_16x16x32_bf16(
                false, a, false, b, (short)0, acc[nn], false, false);
        }
    }

    #pragma unroll
    for (int nn = 0; nn < 4; nn++) {
        #pragma unroll
        for (int r = 0; r < 8; r++) {
            int row = row0 + wm * 16 + r + 8 * hi;
            int c   = wn * 64 + nn * 16 + M;
            float val = (acc[nn][r] + bias[n * DIN + c]) * scale;
            if (OUT_MODE == 0) {
                int t = row >> 3, batch = row & 7;
                int e = n * DIN + c;
                int head = batch * H_DIM + (e >> 6);
                out_bf[((size_t)head * T_DIM + t) * HD + (e & 63)] = (__bf16)val;
            } else {
                out_f[(size_t)row * E_DIM + n * DIN + c] = val;
            }
        }
    }
}

// -----------------------------------------------------------------------------
// Attention with post-softmax top-16.  WG = 1 head x 16 query rows.
// K streamed in 16KB chunks by the TDM into double-buffered LDS while the
// 8 waves compute the previous chunk's 16x16 score tiles with WMMA.
// Top-16 candidate lists live in LDS (one lane owns one row) to keep the
// divergent selection loop off the VGPR budget of the compute waves.
// -----------------------------------------------------------------------------
__global__ __launch_bounds__(256) void attn_kernel(
    const __bf16* __restrict__ qb, const __bf16* __restrict__ kb,
    const __bf16* __restrict__ vb, __bf16* __restrict__ ctx) {
    const int head = blockIdx.y;
    const int q0   = blockIdx.x * 16;
    const int tid  = threadIdx.x;
    const int lane = tid & 31;
    const int wave = tid >> 5;
    const int M    = lane & 15;
    const int hi   = lane >> 4;

    __shared__ __bf16 Ksm[2][CHUNK][HD];   // 32 KB double buffer
    __shared__ float  S[16][CHUNK];        // 8 KB score chunk
    __shared__ float  TopW[16][TOPK];      // raw scores, then weights
    __shared__ int    TopI[16][TOPK];

    const __bf16* khead = kb + (size_t)head * T_DIM * HD;
    const __bf16* vhead = vb + (size_t)head * T_DIM * HD;

    // warm L2 for the data-dependent V gather at the end
    __builtin_prefetch(vhead + (size_t)tid * 256, 0, 1);

    // q fragments (A matrix, 2 K-steps of 32), contiguous head-major loads
    const __bf16* qrowp = qb + ((size_t)head * T_DIM + (q0 + M)) * HD;
    v16bf qf[2];
    #pragma unroll
    for (int s = 0; s < 2; s++) {
        const __bf16* p = qrowp + s * 32 + hi * 16;
        qf[s] = load_frag(p, p + 8);
    }

    // per-row softmax stats in registers; top-16 lists in LDS (lane == row)
    float m = -1e30f, Z = 0.0f;
    float minv = -1e30f;
    int   minp = 0;
    if (tid < 16) {
        #pragma unroll
        for (int i = 0; i < TOPK; i++) { TopW[tid][i] = -1e30f; TopI[tid][i] = 0; }
    }

#if HAS_TDM
    if (tid < 32)  // wave 0 drives the TDM pipeline
        tdm_load_contig(lds_off_of(&Ksm[0][0][0]), khead, CHUNK * HD);
#endif

    for (int c = 0; c < NCHUNK; c++) {
        const int buf = c & 1;
#if HAS_TDM
        if (tid < 32) __builtin_amdgcn_s_wait_tensorcnt((short)0);
        __syncthreads();  // publish chunk c to all waves
        if (c + 1 < NCHUNK && tid < 32)
            tdm_load_contig(lds_off_of(&Ksm[buf ^ 1][0][0]),
                            khead + (size_t)(c + 1) * CHUNK * HD, CHUNK * HD);
#else
        #pragma unroll
        for (int e = 0; e < 4; e++) {   // cooperative 16 KB copy fallback
            int lin = e * 256 + tid;
            ((uint4v*)&Ksm[buf][0][0])[lin] =
                ((const uint4v*)(khead + (size_t)c * CHUNK * HD))[lin];
        }
        __syncthreads();
#endif
        // phase 1: wave w computes score tile for keys [c*CHUNK + w*16 ...)
        {
            v8f acc = {};
            #pragma unroll
            for (int s = 0; s < 2; s++) {
                const __bf16* kp = &Ksm[buf][wave * 16 + M][s * 32 + hi * 16];
                v16bf b = load_frag(kp, kp + 8);
                acc = __builtin_amdgcn_wmma_f32_16x16x32_bf16(
                    false, qf[s], false, b, (short)0, acc, false, false);
            }
            #pragma unroll
            for (int r = 0; r < 8; r++)
                S[r + 8 * hi][wave * 16 + M] = acc[r];
        }
        __syncthreads();

        // phase 2: branchless online softmax + LDS-resident top-16
        if (tid < 16) {
            const int kbase = c * CHUNK;
            for (int j = 0; j < CHUNK; j++) {
                float s  = S[tid][j];
                float nm = fmaxf(m, s);
                Z = Z * __expf(m - nm) + __expf(s - nm);
                m = nm;
                if (s > minv) {                   // rare replacement path
                    TopW[tid][minp] = s;
                    TopI[tid][minp] = kbase + j;
                    minv = TopW[tid][0]; minp = 0;
                    #pragma unroll
                    for (int i = 1; i < TOPK; i++) {
                        float t = TopW[tid][i];
                        if (t < minv) { minv = t; minp = i; }
                    }
                }
            }
        }
        __syncthreads();  // before S / K buffer reuse
    }

    // convert raw scores -> post-softmax weights in place
    if (tid < 16) {
        float inv = 1.0f / Z;
        #pragma unroll
        for (int i = 0; i < TOPK; i++)
            TopW[tid][i] = __expf(TopW[tid][i] - m) * inv;
    }
    __syncthreads();

    // context: 16 rows x 64 dims, 4 dims/thread, 16-key weighted gather of V
    {
        int row = tid >> 4;
        int d0  = (tid & 15) * 4;
        float a0 = 0.f, a1 = 0.f, a2 = 0.f, a3 = 0.f;
        #pragma unroll
        for (int i = 0; i < TOPK; i++) {
            float w  = TopW[row][i];
            int  key = TopI[row][i];
            const __bf16* vp = vhead + (size_t)key * HD + d0;
            a0 += w * (float)vp[0];
            a1 += w * (float)vp[1];
            a2 += w * (float)vp[2];
            a3 += w * (float)vp[3];
        }
        int t = q0 + row;
        int batch = head >> 4;
        int h     = head & 15;
        __bf16* op = ctx + ((size_t)t * B_DIM + batch) * E_DIM + h * HD + d0;
        op[0] = (__bf16)a0; op[1] = (__bf16)a1;
        op[2] = (__bf16)a2; op[3] = (__bf16)a3;
    }
}

// -----------------------------------------------------------------------------
extern "C" void kernel_launch(void* const* d_in, const int* in_sizes, int n_in,
                              void* d_out, int out_size, void* d_ws, size_t ws_size,
                              hipStream_t stream) {
    const float* query  = (const float*)d_in[0];
    const float* key_in = (const float*)d_in[1];
    const float* value  = (const float*)d_in[2];
    const float* Wq = (const float*)d_in[3];
    const float* bq = (const float*)d_in[4];
    const float* Wk = (const float*)d_in[5];
    const float* bk = (const float*)d_in[6];
    const float* Wv = (const float*)d_in[7];
    const float* bv = (const float*)d_in[8];
    const float* Wo = (const float*)d_in[9];
    const float* bo = (const float*)d_in[10];

    char* ws = (char*)d_ws;
    const size_t xbytes = (size_t)ROWS * E_DIM * sizeof(__bf16);       // 16 MiB
    const size_t wbytes = (size_t)NB_DIM * DIN * DIN * sizeof(__bf16); // 512 KiB
    __bf16* xq  = (__bf16*)(ws);                 ws += xbytes;
    __bf16* xk  = (__bf16*)(ws);                 ws += xbytes;
    __bf16* xv  = (__bf16*)(ws);                 ws += xbytes;
    __bf16* wtq = (__bf16*)(ws);                 ws += wbytes;
    __bf16* wtk = (__bf16*)(ws);                 ws += wbytes;
    __bf16* wtv = (__bf16*)(ws);                 ws += wbytes;
    __bf16* wto = (__bf16*)(ws);                 ws += wbytes;
    __bf16* qh  = (__bf16*)(ws);                 ws += xbytes;
    __bf16* kh  = (__bf16*)(ws);                 ws += xbytes;
    __bf16* vh  = (__bf16*)(ws);                 ws += xbytes;
    __bf16* cx  = (__bf16*)(ws);                 ws += xbytes;

    const int n4 = ROWS * E_DIM / 4;
    cvt_bf16_kernel<<<n4 / 256, 256, 0, stream>>>(query,  xq, n4);
    cvt_bf16_kernel<<<n4 / 256, 256, 0, stream>>>(key_in, xk, n4);
    cvt_bf16_kernel<<<n4 / 256, 256, 0, stream>>>(value,  xv, n4);

    dim3 wg(DIN, NB_DIM);
    wtrans_kernel<<<wg, 256, 0, stream>>>(Wq, wtq);
    wtrans_kernel<<<wg, 256, 0, stream>>>(Wk, wtk);
    wtrans_kernel<<<wg, 256, 0, stream>>>(Wv, wtv);
    wtrans_kernel<<<wg, 256, 0, stream>>>(Wo, wto);

    dim3 pg(ROWS / 32, NB_DIM);
    proj_kernel<0><<<pg, 256, 0, stream>>>(xq, wtq, bq, SCALING, qh, nullptr);
    proj_kernel<0><<<pg, 256, 0, stream>>>(xk, wtk, bk, 1.0f,    kh, nullptr);
    proj_kernel<0><<<pg, 256, 0, stream>>>(xv, wtv, bv, 1.0f,    vh, nullptr);

    attn_kernel<<<dim3(T_DIM / 16, NHEAD), 256, 0, stream>>>(qh, kh, vh, cx);

    proj_kernel<1><<<pg, 256, 0, stream>>>(cx, wto, bo, 1.0f, nullptr,
                                           (float*)d_out);
}
